// Attention_15530601742365
// MI455X (gfx1250) — matrix operand order repaired and checked by reading, not compiled, not run on hardware
//
#include <hip/hip_runtime.h>

// ---------- types ----------
typedef _Float16 half16 __attribute__((ext_vector_type(16)));
typedef _Float16 half8  __attribute__((ext_vector_type(8)));
typedef float    v8f    __attribute__((ext_vector_type(8)));
typedef float    f4v    __attribute__((ext_vector_type(4)));
typedef int      i4v    __attribute__((vector_size(16)));

union H16 { half16 v; _Float16 h[16]; half8 h8[2]; };

// Problem constants (B=4, T=4096, C=512, D=64)
#define TT 4096
#define CC 512
#define DD 64
#define BB 4

// ---------- gfx1250 async-copy support (guarded) ----------
#if defined(__AMDGCN__) && __has_builtin(__builtin_amdgcn_global_load_async_to_lds_b128)
#define HAVE_ASYNC 1
#else
#define HAVE_ASYNC 0
#endif

#if HAVE_ASYNC
#if __has_builtin(__builtin_amdgcn_s_wait_asynccnt)
#define WAIT_ASYNC(n) __builtin_amdgcn_s_wait_asynccnt(n)
#else
#define WAIT_ASYNC(n) asm volatile("s_wait_asynccnt " #n)
#endif
typedef __attribute__((address_space(1))) i4v* gvec_p;
typedef __attribute__((address_space(3))) i4v* lvec_p;
__device__ __forceinline__ void async_ld16(const _Float16* g, _Float16* l) {
    void* gp = (void*)g;
    void* lp = (void*)l;
    __builtin_amdgcn_global_load_async_to_lds_b128((gvec_p)gp, (lvec_p)lp, 0, 0);
}
#else
#define WAIT_ASYNC(n)
__device__ __forceinline__ void async_ld16(const _Float16* g, _Float16* l) {
    *(half8*)l = *(const half8*)g;   // synchronous fallback
}
#endif

// ---------- kernel 0: convert + transpose weights to f16 Wt[3][64][512] ----------
__global__ void wt_convert(const float* __restrict__ Wq, const float* __restrict__ Wk,
                           const float* __restrict__ Wv, _Float16* __restrict__ Wt) {
    int tid = blockIdx.x * blockDim.x + threadIdx.x;   // 0 .. 3*32768-1
    int w = tid >> 15;
    int e = tid & 32767;           // e = c*64 + d
    int c = e >> 6;
    int d = e & 63;
    const float* W = (w == 0) ? Wq : ((w == 1) ? Wk : Wv);
    Wt[w * (DD * CC) + d * CC + c] = (_Float16)W[e];
}

// ---------- kernel 1: QKV projection via WMMA ----------
// grid.x = (B*T)/16 row tiles, grid.y = 3 (q,k,v). One wave per tile.
// Q,K stored f16 row-major [B*T][64]; V stored f16 transposed per batch [64][T].
__global__ __launch_bounds__(32) void qkv_proj(
        const float* __restrict__ x, const _Float16* __restrict__ Wt,
        const float* __restrict__ bq, const float* __restrict__ bk, const float* __restrict__ bv,
        _Float16* __restrict__ Qh, _Float16* __restrict__ Kh, _Float16* __restrict__ Vt) {
    const int lane = threadIdx.x & 31;
    const int m  = lane & 15;     // row (A) / column (B) within tile
    const int hi = lane >> 4;
    const int tile  = blockIdx.x;
    const int which = blockIdx.y;
    const int r0 = tile * 16;
    const _Float16* Wm   = Wt + which * (DD * CC);
    const float*    bias = (which == 0) ? bq : ((which == 1) ? bk : bv);

    v8f acc[4];
    for (int t = 0; t < 4; ++t)
        for (int i = 0; i < 8; ++i) acc[t][i] = 0.f;

    const float* xrow = x + (size_t)(r0 + m) * CC;
    for (int kc = 0; kc < CC; kc += 32) {
        H16 a;
        const f4v* p0 = (const f4v*)(xrow + kc + 8 * hi);
        f4v f0 = p0[0], f1 = p0[1];
        const f4v* p1 = (const f4v*)(xrow + kc + 16 + 8 * hi);
        f4v f2 = p1[0], f3 = p1[1];
        for (int j = 0; j < 4; ++j) {
            a.h[j]      = (_Float16)f0[j];
            a.h[4 + j]  = (_Float16)f1[j];
            a.h[8 + j]  = (_Float16)f2[j];
            a.h[12 + j] = (_Float16)f3[j];
        }
        for (int nt = 0; nt < 4; ++nt) {
            H16 bf;
            const half8* wp = (const half8*)(Wm + (size_t)(nt * 16 + m) * CC + kc + 8 * hi);
            bf.h8[0] = wp[0];
            bf.h8[1] = wp[2];
            acc[nt] = __builtin_amdgcn_wmma_f32_16x16x32_f16(
                false, a.v, false, bf.v, (short)0, acc[nt], false, false);
        }
    }
    for (int nt = 0; nt < 4; ++nt) {
        float bvv = bias[nt * 16 + m];
        for (int i = 0; i < 8; ++i) acc[nt][i] += bvv;
    }
    if (which < 2) {
        _Float16* Out = (which == 0) ? Qh : Kh;
        for (int nt = 0; nt < 4; ++nt)
            for (int i = 0; i < 8; ++i) {
                int row = i + 8 * hi;
                Out[(size_t)(r0 + row) * DD + nt * 16 + m] = (_Float16)acc[nt][i];
            }
    } else {
        int b   = r0 >> 12;
        int tl0 = r0 & (TT - 1);
        _Float16* Vb = Vt + (size_t)b * DD * TT;
        for (int nt = 0; nt < 4; ++nt)
            for (int i = 0; i < 8; ++i) {
                int row = i + 8 * hi;
                Vb[(size_t)(nt * 16 + m) * TT + tl0 + row] = (_Float16)acc[nt][i];
            }
    }
}

// ---------- kernel 2: causal flash attention ----------
// Block = 128 threads (4 waves), each wave owns a 16-query tile of a 64-query
// superblock. K/V 32-key stages are double-buffered in LDS via async copies.
__global__ __launch_bounds__(128) void attn(
        const _Float16* __restrict__ Qh, const _Float16* __restrict__ Kh,
        const _Float16* __restrict__ Vt, float* __restrict__ Out) {
    __shared__ _Float16 Kl[2][32][64];   // keys x d      (8 KB)
    __shared__ _Float16 Vl[2][64][32];   // d x keys      (8 KB)
    __shared__ _Float16 Pw[4][16][32];   // per-wave P    (4 KB)

    const int tid  = threadIdx.x;
    const int w    = tid >> 5;          // wave id 0..3
    const int lane = tid & 31;
    const int n  = lane & 15;
    const int hi = lane >> 4;
    const int b  = blockIdx.x >> 6;     // / (T/64)
    const int qb = blockIdx.x & 63;
    const int q0  = qb * 64;            // superblock base
    const int q0w = q0 + 16 * w;        // this wave's query tile
    const _Float16* Qb = Qh + (size_t)b * TT * DD;
    const _Float16* Kb = Kh + (size_t)b * TT * DD;
    const _Float16* Vb = Vt + (size_t)b * DD * TT;

    // staging addressing (per thread, 4x b128 per stage)
    const int kr = tid >> 2, ks = (tid & 3) * 16;   // K: row 0..31, 16-half seg
    const int vr = tid >> 1, vs = (tid & 1) * 16;   // V: row 0..63, 16-half seg

    // Q A-fragments (two 32-wide D chunks), lane row = n
    H16 qa[2];
    {
        const _Float16* qrow = Qb + (size_t)(q0w + n) * DD;
        for (int dc = 0; dc < 2; ++dc) {
            qa[dc].h8[0] = *(const half8*)(qrow + dc * 32 + 8 * hi);
            qa[dc].h8[1] = *(const half8*)(qrow + dc * 32 + 16 + 8 * hi);
        }
    }

    float mr[8], lr[8];
    v8f o[4];
    for (int i = 0; i < 8; ++i) { mr[i] = -1e30f; lr[i] = 0.f; }
    for (int t = 0; t < 4; ++t)
        for (int i = 0; i < 8; ++i) o[t][i] = 0.f;

    const int nblkmax = (q0 + 64 + 31) >> 5;   // stages for the whole superblock
    const int nblkw   = (q0w + 16 + 31) >> 5;  // stages this wave computes

    // prime stage 0
    {
        const int s0 = 0;
        async_ld16(Kb + (size_t)(s0 + kr) * DD + ks,     &Kl[0][kr][ks]);
        async_ld16(Kb + (size_t)(s0 + kr) * DD + ks + 8, &Kl[0][kr][ks + 8]);
        async_ld16(Vb + (size_t)vr * TT + s0 + vs,       &Vl[0][vr][vs]);
        async_ld16(Vb + (size_t)vr * TT + s0 + vs + 8,   &Vl[0][vr][vs + 8]);
    }

    for (int blk = 0; blk < nblkmax; ++blk) {
        const int s0  = blk * 32;
        const int buf = blk & 1;
        const bool more = (blk + 1) < nblkmax;
        if (more) {                      // prefetch next stage into other buffer
            const int s1 = s0 + 32, nb = buf ^ 1;
            async_ld16(Kb + (size_t)(s1 + kr) * DD + ks,     &Kl[nb][kr][ks]);
            async_ld16(Kb + (size_t)(s1 + kr) * DD + ks + 8, &Kl[nb][kr][ks + 8]);
            async_ld16(Vb + (size_t)vr * TT + s1 + vs,       &Vl[nb][vr][vs]);
            async_ld16(Vb + (size_t)vr * TT + s1 + vs + 8,   &Vl[nb][vr][vs + 8]);
            WAIT_ASYNC(4);               // current stage complete (in-order)
        } else {
            WAIT_ASYNC(0);
        }
        __syncthreads();

        if (blk < nblkw) {               // wave-uniform: EXEC stays all-1s
            // ---- S = Q * K^T from LDS ----
            v8f S[2];
            for (int kt = 0; kt < 2; ++kt) {
                v8f c;
                for (int i = 0; i < 8; ++i) c[i] = 0.f;
                const _Float16* krow = &Kl[buf][kt * 16 + n][0];
                for (int dc = 0; dc < 2; ++dc) {
                    H16 kf;
                    const half8* p = (const half8*)(krow + dc * 32 + 8 * hi);
                    kf.h8[0] = p[0];
                    kf.h8[1] = p[2];
                    c = __builtin_amdgcn_wmma_f32_16x16x32_f16(
                        false, qa[dc].v, false, kf.v, (short)0, c, false, false);
                }
                S[kt] = c;
            }
            // ---- scale (*sqrt(D)=8), causal mask, online softmax ----
            const bool maskblk = (s0 + 31 > q0w);
            for (int i = 0; i < 8; ++i) {
                int row = i + 8 * hi;
                float x0 = S[0][i] * 8.0f;
                float x1 = S[1][i] * 8.0f;
                if (maskblk) {
                    if (s0 + n      > q0w + row) x0 = -1e30f;
                    if (s0 + 16 + n > q0w + row) x1 = -1e30f;
                }
                float mx = fmaxf(x0, x1);
                mx = fmaxf(mx, __shfl_xor(mx, 1, 32));
                mx = fmaxf(mx, __shfl_xor(mx, 2, 32));
                mx = fmaxf(mx, __shfl_xor(mx, 4, 32));
                mx = fmaxf(mx, __shfl_xor(mx, 8, 32));
                float mnew  = fmaxf(mr[i], mx);
                float alpha = __expf(mr[i] - mnew);
                float p0 = __expf(x0 - mnew);
                float p1 = __expf(x1 - mnew);
                float ps = p0 + p1;
                ps += __shfl_xor(ps, 1, 32);
                ps += __shfl_xor(ps, 2, 32);
                ps += __shfl_xor(ps, 4, 32);
                ps += __shfl_xor(ps, 8, 32);
                lr[i] = lr[i] * alpha + ps;
                mr[i] = mnew;
                for (int t = 0; t < 4; ++t) o[t][i] *= alpha;
                Pw[w][row][n]      = (_Float16)p0;
                Pw[w][row][16 + n] = (_Float16)p1;
            }
            // ---- P A-fragment (per-wave LDS; compiler orders ds RAW) ----
            H16 pa;
            {
                const _Float16* prow = &Pw[w][n][0];
                pa.h8[0] = *(const half8*)(prow + 8 * hi);
                pa.h8[1] = *(const half8*)(prow + 16 + 8 * hi);
            }
            // ---- O += P * V from LDS ----
            for (int nt = 0; nt < 4; ++nt) {
                H16 vf;
                const half8* p = (const half8*)(&Vl[buf][nt * 16 + n][0] + 8 * hi);
                vf.h8[0] = p[0];
                vf.h8[1] = p[2];
                o[nt] = __builtin_amdgcn_wmma_f32_16x16x32_f16(
                    false, pa.v, false, vf.v, (short)0, o[nt], false, false);
            }
        }
        __syncthreads();                 // protect buffer reuse next iteration
    }

    // ---- normalize and store fp32 ----
    float* Ob = Out + (size_t)b * TT * DD;
    for (int nt = 0; nt < 4; ++nt)
        for (int i = 0; i < 8; ++i) {
            int row = i + 8 * hi;
            Ob[(size_t)(q0w + row) * DD + nt * 16 + n] = o[nt][i] / lr[i];
        }
}

// ---------- launcher ----------
extern "C" void kernel_launch(void* const* d_in, const int* in_sizes, int n_in,
                              void* d_out, int out_size, void* d_ws, size_t ws_size,
                              hipStream_t stream) {
    const float* x  = (const float*)d_in[0];
    const float* Wq = (const float*)d_in[1];
    const float* bq = (const float*)d_in[2];
    const float* Wk = (const float*)d_in[3];
    const float* bk = (const float*)d_in[4];
    const float* Wv = (const float*)d_in[5];
    const float* bv = (const float*)d_in[6];
    float* out = (float*)d_out;

    char* ws = (char*)d_ws;
    const size_t WT_BYTES  = 3ull * DD * CC * sizeof(_Float16);
    const size_t QKV_BYTES = (size_t)BB * TT * DD * sizeof(_Float16);
    _Float16* Wt = (_Float16*)ws;
    _Float16* Qh = (_Float16*)(ws + WT_BYTES);
    _Float16* Kh = (_Float16*)(ws + WT_BYTES + QKV_BYTES);
    _Float16* Vt = (_Float16*)(ws + WT_BYTES + 2 * QKV_BYTES);

    wt_convert<<<dim3((3 * CC * DD) / 256), 256, 0, stream>>>(Wq, Wk, Wv, Wt);
    qkv_proj<<<dim3((BB * TT) / 16, 3), 32, 0, stream>>>(x, Wt, bq, bk, bv, Qh, Kh, Vt);
    attn<<<dim3((BB * TT) / 64), 128, 0, stream>>>(Qh, Kh, Vt, out);
}
